// BiDecoderRNN_39513699123568
// MI455X (gfx1250) — matrix operand order repaired and checked by reading, not compiled, not run on hardware
//
#include <hip/hip_runtime.h>
#include <hip/hip_bf16.h>
#include <math.h>

// ---------------- problem constants ----------------
#define HD   256
#define IND  128
#define AD   64
#define FD   1024
#define VD   50000
#define SD   512
#define TD   32
#define BD   32

typedef __bf16 bf16_t;
typedef __attribute__((ext_vector_type(16))) __bf16 v16bf;
typedef __attribute__((ext_vector_type(8)))  __bf16 v8bf;
typedef __attribute__((ext_vector_type(8)))  float  v8f;

__device__ __forceinline__ float sigm_(float x) { return 1.0f / (1.0f + __expf(-x)); }

// ======================================================================
// bf16 WMMA GEMM:  C(M,N) = A(M,K) @ W(N,K)^T + bias,  f32 accum.
// Each wave computes a 32x16 tile (two 16x16 WMMA tiles sharing one
// B-fragment) -> every W element is loaded ONCE per GEMM, which matters
// because the dominant GEMM (M=32, N=50000, K=1024) is bound by W_v2
// bandwidth (bf16 W_v2 = 102.4 MB, L2-resident across the 32 timesteps).
// Requires M%32==0, N%16==0, K%32==0 (true for every GEMM here).
//
// A-frag (16x32 bf16): lane l holds row m=l&15; lanes 0-15 K={k0..k0+7,
// k0+16..k0+23}, lanes 16-31 K shifted by +8  -> two 16B loads per lane.
// B-frag (32x16): lane l holds col n=l&15; same K order as A-frag
// -> two 16B loads per lane.
// ======================================================================
__global__ void k_gemm_bf16(const bf16_t* __restrict__ A,
                            const bf16_t* __restrict__ W,
                            const float*  __restrict__ bias,
                            float*        __restrict__ C,
                            bf16_t*       __restrict__ Cbf,
                            int M, int N, int K) {
  const int gtid  = blockIdx.x * blockDim.x + threadIdx.x;
  const int wave  = gtid >> 5;
  const int lane  = gtid & 31;
  const int ntile = N >> 4;
  const int mgrp  = M >> 5;                      // 32-row groups
  if (wave >= mgrp * ntile) return;
  const int mg = wave / ntile;
  const int nt = wave - mg * ntile;
  const int l15   = lane & 15;
  const int khalf = lane >> 4;

  const size_t arow0 = (size_t)(mg * 32 + l15) * (size_t)K;        // m-tile 0
  const size_t arow1 = arow0 + (size_t)16 * (size_t)K;             // m-tile 1
  const size_t brow  = (size_t)(nt * 16 + l15) * (size_t)K;

  v8f acc0 = {};
  v8f acc1 = {};
  for (int k0 = 0; k0 < K; k0 += 32) {
    union { v16bf v; v8bf h[2]; } a0, a1, b;
    a0.h[0] = *(const v8bf*)(A + arow0 + k0 + khalf * 8);
    a0.h[1] = *(const v8bf*)(A + arow0 + k0 + 16 + khalf * 8);
    a1.h[0] = *(const v8bf*)(A + arow1 + k0 + khalf * 8);
    a1.h[1] = *(const v8bf*)(A + arow1 + k0 + 16 + khalf * 8);
    const v8bf* pb = (const v8bf*)(W + brow + k0 + khalf * 8);
    b.h[0] = pb[0];
    b.h[1] = pb[2];
    // prefetch next K-tile of the (streaming) weight operand
    __builtin_prefetch(W + brow + k0 + 32 + khalf * 16, 0, 1);
    acc0 = __builtin_amdgcn_wmma_f32_16x16x32_bf16(
        false, a0.v, false, b.v, (short)0, acc0, false, false);
    acc1 = __builtin_amdgcn_wmma_f32_16x16x32_bf16(
        false, a1.v, false, b.v, (short)0, acc1, false, false);
  }

  const int nn = nt * 16 + l15;
  const float bv = bias ? bias[nn] : 0.0f;
#pragma unroll
  for (int r = 0; r < 8; ++r) {                  // row m = base + khalf*8 + r
    const int mm0 = mg * 32 + khalf * 8 + r;
    const int mm1 = mm0 + 16;
    const float v0 = acc0[r] + bv;
    const float v1 = acc1[r] + bv;
    C[(size_t)mm0 * N + nn] = v0;
    C[(size_t)mm1 * N + nn] = v1;
    if (Cbf) {
      Cbf[(size_t)mm0 * N + nn] = (bf16_t)v0;
      Cbf[(size_t)mm1 * N + nn] = (bf16_t)v1;
    }
  }
}

// ---------------- f32 -> bf16 strided convert (grid-stride) ----------------
__global__ void k_cvt_bf16(const float* __restrict__ src, long ss, long soff,
                           bf16_t* __restrict__ dst, long ds, long doff,
                           long rows, long cols) {
  const long total = rows * cols;
  for (long i = (long)blockIdx.x * blockDim.x + threadIdx.x; i < total;
       i += (long)gridDim.x * blockDim.x) {
    const long r = i / cols, c = i - r * cols;
    dst[r * ds + doff + c] = (bf16_t)src[r * ss + soff + c];
  }
}

// ---------------- bcat = b_ih + b_hh ----------------
__global__ void k_addvec(const float* __restrict__ a, const float* __restrict__ b,
                         float* __restrict__ o, int n) {
  int i = blockIdx.x * blockDim.x + threadIdx.x;
  if (i < n) o[i] = a[i] + b[i];
}

// ---------------- old_enc = concat(src[0:2], src[2:4], axis=-1) -> bf16 ----
__global__ void k_build_old(const float* __restrict__ src4, bf16_t* __restrict__ dst) {
  const int total = 2 * BD * 2 * HD;
  int i = blockIdx.x * blockDim.x + threadIdx.x;
  if (i >= total) return;
  const int row = i / (2 * HD);          // l*BD + b
  const int k   = i - row * (2 * HD);
  const int l   = row / BD, b = row - l * BD;
  const float v = (k < HD) ? src4[(size_t)(l * BD + b) * HD + k]
                           : src4[(size_t)((l + 2) * BD + b) * HD + (k - HD)];
  dst[i] = (bf16_t)v;
}

// ---------------- LSTM pointwise: gates -> (c,h), fused bf16 re-pack ------
__global__ void k_lstm_pw(const float* __restrict__ g,
                          float* __restrict__ c, float* __restrict__ h,
                          bf16_t* d1, int s1, int o1,
                          bf16_t* d2, int s2, int o2,
                          bf16_t* d3, int s3, int o3) {
  int idx = blockIdx.x * blockDim.x + threadIdx.x;
  if (idx >= BD * HD) return;
  const int b = idx / HD, j = idx - b * HD;
  const float* gb = g + (size_t)b * 4 * HD;
  const float ig = sigm_(gb[j]);
  const float fg = sigm_(gb[HD + j]);
  const float gg = tanhf(gb[2 * HD + j]);
  const float og = sigm_(gb[3 * HD + j]);
  const float cn = fg * c[idx] + ig * gg;
  const float hn = og * tanhf(cn);
  c[idx] = cn;
  h[idx] = hn;
  const bf16_t hb = (bf16_t)hn;
  if (d1) d1[(size_t)b * s1 + o1 + j] = hb;
  if (d2) d2[(size_t)b * s2 + o2 + j] = hb;
  if (d3) d3[(size_t)b * s3 + o3 + j] = hb;
}

// ---------------- attention scalar score per (s,b) ----------------
__global__ void k_att_escore(const float* __restrict__ encproj,   // (S*B, A)
                             const float* __restrict__ decproj,   // (B, A)
                             const float* __restrict__ coverage,  // (S*B)
                             const float* __restrict__ W_att,     // (A, 3H+1)
                             const float* __restrict__ W_attout,  // (A)
                             const float* __restrict__ att_mask,  // (S*B)
                             float* __restrict__ esc) {
  int idx = blockIdx.x * blockDim.x + threadIdx.x;
  if (idx >= SD * BD) return;
  const int b = idx & (BD - 1);
  const float cov = coverage[idx];
  const float* ep = encproj + (size_t)idx * AD;
  float acc = 0.0f;
#pragma unroll 4
  for (int a = 0; a < AD; ++a) {
    const float e = tanhf(ep[a] + decproj[b * AD + a] + cov * W_att[a * (3 * HD + 1) + 3 * HD]);
    acc += e * W_attout[a];
  }
  esc[idx] = acc + att_mask[idx];
}

// ---------------- softmax over S + coverage update + cov_loss ----------------
__global__ void k_att_softmax(const float* __restrict__ esc,
                              float* __restrict__ coverage,
                              float* __restrict__ attn,
                              float* __restrict__ covloss) {   // covloss[b]
  __shared__ float red[256];
  const int b = blockIdx.x;
  const int tid = threadIdx.x;
  float mx = -1e30f;
  for (int s = tid; s < SD; s += 256) mx = fmaxf(mx, esc[s * BD + b]);
  red[tid] = mx; __syncthreads();
  for (int o = 128; o > 0; o >>= 1) { if (tid < o) red[tid] = fmaxf(red[tid], red[tid + o]); __syncthreads(); }
  mx = red[0]; __syncthreads();
  float sum = 0.0f;
  for (int s = tid; s < SD; s += 256) sum += __expf(esc[s * BD + b] - mx);
  red[tid] = sum; __syncthreads();
  for (int o = 128; o > 0; o >>= 1) { if (tid < o) red[tid] += red[tid + o]; __syncthreads(); }
  sum = red[0]; __syncthreads();
  const float inv = 1.0f / sum;
  float closs = 0.0f;
  for (int s = tid; s < SD; s += 256) {
    const float a  = __expf(esc[s * BD + b] - mx) * inv;
    const float cv = coverage[s * BD + b];
    closs += fminf(a, cv);
    coverage[s * BD + b] = cv + a;
    attn[s * BD + b] = a;
  }
  red[tid] = closs; __syncthreads();
  for (int o = 128; o > 0; o >>= 1) { if (tid < o) red[tid] += red[tid + o]; __syncthreads(); }
  if (tid == 0) covloss[b] = red[0];
}

// ---------------- context = sum_s attn * output_enc -> bf16 into pv_in ------
__global__ void k_att_context(const float* __restrict__ attn,
                              const float* __restrict__ output_enc,  // (S,B,2H)
                              bf16_t* __restrict__ pvin) {           // (B,3H)
  int idx = blockIdx.x * blockDim.x + threadIdx.x;   // b*2H + d
  if (idx >= BD * 2 * HD) return;
  const int b = idx >> 9;
  const int d = idx & (2 * HD - 1);
  float acc = 0.0f;
  for (int s = 0; s < SD; ++s)
    acc = fmaf(attn[s * BD + b], output_enc[((size_t)s * BD + b) * (2 * HD) + d], acc);
  pvin[(size_t)b * (3 * HD) + HD + d] = (bf16_t)acc;
}

// ======================================================================
extern "C" void kernel_launch(void* const* d_in, const int* in_sizes, int n_in,
                              void* d_out, int out_size, void* d_ws, size_t ws_size,
                              hipStream_t stream) {
  (void)in_sizes; (void)n_in; (void)out_size; (void)ws_size;
  const float* output_enc = (const float*)d_in[0];
  const float* input_dec  = (const float*)d_in[1];
  const float* hidden_enc = (const float*)d_in[2];
  const float* cell_enc   = (const float*)d_in[3];
  const float* att_mask   = (const float*)d_in[4];
  const float* W_ih0 = (const float*)d_in[5];
  const float* W_hh0 = (const float*)d_in[6];
  const float* b_ih0 = (const float*)d_in[7];
  const float* b_hh0 = (const float*)d_in[8];
  const float* W_ih1 = (const float*)d_in[9];
  const float* W_hh1 = (const float*)d_in[10];
  const float* b_ih1 = (const float*)d_in[11];
  const float* b_hh1 = (const float*)d_in[12];
  const float* W_red = (const float*)d_in[13];
  const float* b_red = (const float*)d_in[14];
  const float* W_att = (const float*)d_in[15];
  const float* b_att = (const float*)d_in[16];
  const float* W_attout = (const float*)d_in[17];
  const float* W_v1 = (const float*)d_in[18];
  const float* b_v1 = (const float*)d_in[19];
  const float* W_v2 = (const float*)d_in[20];
  const float* b_v2 = (const float*)d_in[21];
  float* fout = (float*)d_out;

  // ---- workspace bump allocator ----
  char* ws = (char*)d_ws;
  size_t off = 0;
  auto alloc = [&](size_t bytes) -> void* {
    void* p = ws + off;
    off = (off + bytes + 255) & ~(size_t)255;
    return p;
  };
  // bf16 buffers
  bf16_t* wcat0  = (bf16_t*)alloc((size_t)4 * HD * (IND + HD) * 2);        // (1024,384)
  bf16_t* wcat1  = (bf16_t*)alloc((size_t)4 * HD * (2 * HD) * 2);          // (1024,512)
  bf16_t* wredb  = (bf16_t*)alloc((size_t)HD * 2 * HD * 2);                // (256,512)
  bf16_t* waenc  = (bf16_t*)alloc((size_t)AD * 2 * HD * 2);                // (64,512)
  bf16_t* wadec  = (bf16_t*)alloc((size_t)AD * HD * 2);                    // (64,256)
  bf16_t* wv1b   = (bf16_t*)alloc((size_t)FD * 3 * HD * 2);                // (1024,768)
  bf16_t* wv2b   = (bf16_t*)alloc((size_t)VD * FD * 2);                    // 102.4 MB (L2-resident)
  bf16_t* encA   = (bf16_t*)alloc((size_t)SD * BD * 2 * HD * 2);           // (16384,512)
  bf16_t* oldenc = (bf16_t*)alloc((size_t)2 * BD * 2 * HD * 2);
  bf16_t* oldcel = (bf16_t*)alloc((size_t)2 * BD * 2 * HD * 2);
  bf16_t* xcat0  = (bf16_t*)alloc((size_t)BD * (IND + HD) * 2);            // (32,384)
  bf16_t* xcat1  = (bf16_t*)alloc((size_t)BD * (2 * HD) * 2);              // (32,512)
  bf16_t* decbf  = (bf16_t*)alloc((size_t)BD * HD * 2);                    // (32,256)
  bf16_t* pvin   = (bf16_t*)alloc((size_t)BD * 3 * HD * 2);                // (32,768)
  bf16_t* pv1b   = (bf16_t*)alloc((size_t)BD * FD * 2);                    // (32,1024)
  // f32 buffers
  float* encproj = (float*)alloc((size_t)SD * BD * AD * 4);                // 4 MB
  float* henc    = (float*)alloc((size_t)2 * BD * HD * 4);                 // states h0|h1
  float* cenc    = (float*)alloc((size_t)2 * BD * HD * 4);                 // states c0|c1
  float* bcat0   = (float*)alloc((size_t)4 * HD * 4);
  float* bcat1   = (float*)alloc((size_t)4 * HD * 4);
  float* gates   = (float*)alloc((size_t)BD * 4 * HD * 4);
  float* decproj = (float*)alloc((size_t)BD * AD * 4);
  float* esc     = (float*)alloc((size_t)SD * BD * 4);
  float* cover   = (float*)alloc((size_t)SD * BD * 4);
  float* attn    = (float*)alloc((size_t)SD * BD * 4);
  float* pv1f    = (float*)alloc((size_t)BD * FD * 4);

  auto cvt = [&](const float* src, long ss, long soff, bf16_t* dst, long ds, long doff,
                 long rows, long cols) {
    long total = rows * cols;
    int grid = (int)((total + 255) / 256);
    k_cvt_bf16<<<grid, 256, 0, stream>>>(src, ss, soff, dst, ds, doff, rows, cols);
  };
  auto gemm = [&](const bf16_t* A, const bf16_t* W, const float* bias,
                  float* C, bf16_t* Cbf, int M, int N, int K) {
    long tiles = (long)(M / 32) * (N / 16);      // one wave per 32x16 tile
    int grid = (int)((tiles * 32 + 255) / 256);
    k_gemm_bf16<<<grid, 256, 0, stream>>>(A, W, bias, C, Cbf, M, N, K);
  };

  // ---- one-time (per call) weight/activation conversion ----
  cvt(W_ih0, IND, 0, wcat0, IND + HD, 0,   4 * HD, IND);     // [W_ih0 | W_hh0]
  cvt(W_hh0, HD,  0, wcat0, IND + HD, IND, 4 * HD, HD);
  cvt(W_ih1, HD,  0, wcat1, 2 * HD, 0,  4 * HD, HD);         // [W_ih1 | W_hh1]
  cvt(W_hh1, HD,  0, wcat1, 2 * HD, HD, 4 * HD, HD);
  cvt(W_red, 2 * HD, 0, wredb, 2 * HD, 0, HD, 2 * HD);
  cvt(W_att, 3 * HD + 1, 0,      waenc, 2 * HD, 0, AD, 2 * HD);   // Wa_enc
  cvt(W_att, 3 * HD + 1, 2 * HD, wadec, HD,     0, AD, HD);       // Wa_dec
  cvt(W_v1, 3 * HD, 0, wv1b, 3 * HD, 0, FD, 3 * HD);
  cvt(W_v2, FD, 0, wv2b, FD, 0, VD, FD);                     // 51.2M elems -> bf16 (L2-resident)
  cvt(output_enc, 2 * HD, 0, encA, 2 * HD, 0, (long)SD * BD, 2 * HD);
  k_addvec<<<(4 * HD + 255) / 256, 256, 0, stream>>>(b_ih0, b_hh0, bcat0, 4 * HD);
  k_addvec<<<(4 * HD + 255) / 256, 256, 0, stream>>>(b_ih1, b_hh1, bcat1, 4 * HD);
  k_build_old<<<(2 * BD * 2 * HD + 255) / 256, 256, 0, stream>>>(hidden_enc, oldenc);
  k_build_old<<<(2 * BD * 2 * HD + 255) / 256, 256, 0, stream>>>(cell_enc, oldcel);

  // ---- initial states: new_enc/new_cell = old @ W_red.T + b_red ----
  gemm(oldenc, wredb, b_red, henc, nullptr, 2 * BD, HD, 2 * HD);
  gemm(oldcel, wredb, b_red, cenc, nullptr, 2 * BD, HD, 2 * HD);
  // enc_proj = output_enc @ Wa_enc.T + b_att   (S*B, A)
  gemm(encA, waenc, b_att, encproj, nullptr, SD * BD, AD, 2 * HD);

  // seed recurrent bf16 inputs: h0 -> xcat0[:,128:], h1 -> xcat1[:,256:]
  cvt(henc, HD, 0, xcat0, IND + HD, IND, BD, HD);
  cvt(henc + (size_t)BD * HD, HD, 0, xcat1, 2 * HD, HD, BD, HD);
  hipMemsetAsync(cover, 0, (size_t)SD * BD * 4, stream);

  float* h0 = henc;                   float* c0 = cenc;
  float* h1 = henc + (size_t)BD * HD; float* c1 = cenc + (size_t)BD * HD;
  const size_t covlossOff = (size_t)TD * BD * VD;
  const size_t hiddenOff  = covlossOff + (size_t)TD * BD;
  const size_t cellOff    = hiddenOff + (size_t)2 * BD * HD;

  // ---- sequential scan over T timesteps (LSTM + attention interleaved) ----
  for (int t = 0; t < TD; ++t) {
    // x_t -> bf16 into xcat0[:, 0:128]
    cvt(input_dec + (size_t)t * BD * IND, IND, 0, xcat0, IND + HD, 0, BD, IND);
    // layer 0: gates = [x,h0] @ [W_ih0|W_hh0]^T + bcat0
    gemm(xcat0, wcat0, bcat0, gates, nullptr, BD, 4 * HD, IND + HD);
    k_lstm_pw<<<(BD * HD + 255) / 256, 256, 0, stream>>>(
        gates, c0, h0,
        xcat0, IND + HD, IND,     // recurrent input of layer 0 (next step)
        xcat1, 2 * HD, 0,         // y0_t = input of layer 1 (this step)
        nullptr, 0, 0);
    // layer 1
    gemm(xcat1, wcat1, bcat1, gates, nullptr, BD, 4 * HD, 2 * HD);
    k_lstm_pw<<<(BD * HD + 255) / 256, 256, 0, stream>>>(
        gates, c1, h1,
        xcat1, 2 * HD, HD,        // recurrent input of layer 1 (next step)
        decbf, HD, 0,             // dec_t bf16 for attention
        pvin, 3 * HD, 0);         // dec part of pv_in
    // attention
    gemm(decbf, wadec, nullptr, decproj, nullptr, BD, AD, HD);
    k_att_escore<<<(SD * BD + 255) / 256, 256, 0, stream>>>(
        encproj, decproj, cover, W_att, W_attout, att_mask, esc);
    k_att_softmax<<<BD, 256, 0, stream>>>(esc, cover, attn, fout + covlossOff + (size_t)t * BD);
    k_att_context<<<(BD * 2 * HD + 255) / 256, 256, 0, stream>>>(attn, output_enc, pvin);
    // vocabulary projection: pv1 = pv_in @ W_v1^T + b_v1 ; pvocab = pv1 @ W_v2^T + b_v2
    gemm(pvin, wv1b, b_v1, pv1f, pv1b, BD, FD, 3 * HD);
    gemm(pv1b, wv2b, b_v2, fout + (size_t)t * BD * VD, nullptr, BD, VD, FD);
  }

  // ---- final hidden/cell states (h0|h1 and c0|c1 are contiguous) ----
  hipMemcpyAsync(fout + hiddenOff, henc, (size_t)2 * BD * HD * 4,
                 hipMemcpyDeviceToDevice, stream);
  hipMemcpyAsync(fout + cellOff, cenc, (size_t)2 * BD * HD * 4,
                 hipMemcpyDeviceToDevice, stream);
}
